// SSMFoundationalDecoder_27487790694540
// MI455X (gfx1250) — matrix operand order repaired and checked by reading, not compiled, not run on hardware
//
#include <hip/hip_runtime.h>
#include <math.h>

#define T_LEN 131072
#define IN_DIM 256
#define H_DIM 64
#define P_DIM 32
#define NL 4
#define OUT_DIM 2
#define CHUNK 512
#define NCHUNK (T_LEN / CHUNK)   // 256
#define ROWS 128                 // rows per GEMM block (8 waves x 16-row M tiles)
#define LDT 72                   // LDS tile row stride (halfs), padded

typedef __attribute__((ext_vector_type(16))) _Float16 v16h;
typedef __attribute__((ext_vector_type(8)))  float    v8f;

// ---------------------------------------------------------------- WMMA helpers
__device__ inline v8f wmma_f16(v16h a, v16h b, v8f c) {
  // D = A(16x32 f16) * B(32x16 f16) + C(16x16 f32)
  return __builtin_amdgcn_wmma_f32_16x16x32_f16(false, a, false, b, (short)0, c,
                                                false, false);
}

// A-fragment: 16x32 f16, contiguous k order (k = 16*(lane>>4) + s); contiguous halves -> 2x ds_load_b128.
__device__ inline v16h load_a_frag(const _Float16* base, int ld, int koff) {
  int lane = threadIdx.x & 31;
  const _Float16* p = base + (lane & 15) * ld + koff + ((lane >> 4) * 16);
  v16h a;
#pragma unroll
  for (int v = 0; v < 4; ++v) {
    a[2 * v]     = p[2 * v];
    a[2 * v + 1] = p[2 * v + 1];
    a[2 * v + 8] = p[8 + 2 * v];
    a[2 * v + 9] = p[8 + 2 * v + 1];
  }
  return a;
}

// B-fragment from a pre-swizzled buffer: lane l's 16 halves are contiguous
// at frag*512 + l*16  ->  one aligned 32B vector load (2x global_load_b128).
__device__ inline v16h load_b_frag_swz(const _Float16* __restrict__ Bs, int frag) {
  int lane = threadIdx.x & 31;
  return *(const v16h*)(Bs + (size_t)frag * 512 + lane * 16);
}

__device__ inline float gelu_f(float x) {
  return 0.5f * x * (1.f + tanhf(0.7978845608028654f * (x + 0.044715f * x * x * x)));
}
__device__ inline float sigmoid_f(float x) { return 1.f / (1.f + expf(-x)); }

// ---------------------------------------------------------------- prep kernel
// Emits every weight matrix directly in swizzled WMMA B-fragment order:
//   element e of lane l in fragment (ks, nt) = B[ks*32 + (l>>4)*16 + e][nt*16 + (l&15)]
__global__ void prep_kernel(const int* gidx, const float* enc_w, const float* enc_b,
                            const float* ctx_emb, const float* Lre, const float* Lim,
                            const float* B_re, const float* B_im,
                            const float* C_re, const float* C_im, const float* log_step,
                            const float* glu_w1, const float* glu_w2,
                            _Float16* encBs, float* b64, _Float16* Bcats, _Float16* WcBs,
                            _Float16* w1Bs, _Float16* w2Bs, float* LbarT) {
  __shared__ float cre[NL][32], cim[NL][32];
  int g = gidx[0];
  int tid = threadIdx.x;

  if (tid < 64)
    b64[tid] = (tid < 56) ? enc_b[g * 56 + tid] : ctx_emb[g * 8 + (tid - 56)];

  if (tid < 32) {
    for (int l = 0; l < NL; ++l) {
      float ar = Lre[l * 32 + tid], ai = Lim[l * 32 + tid];
      float dt = expf(log_step[l * 32 + tid]);
      float er = expf(ar * dt);
      float lr = er * cosf(ai * dt), li = er * sinf(ai * dt);
      LbarT[l * 64 + tid] = lr;
      LbarT[l * 64 + 32 + tid] = li;
      float nr = lr - 1.f, ni = li;
      float d2 = ar * ar + ai * ai;
      cre[l][tid] = (nr * ar + ni * ai) / d2;   // (Lbar-1)/Lambda
      cim[l][tid] = (ni * ar - nr * ai) / d2;
    }
  }
  __syncthreads();

  // Encoder weights: 8 k-steps x 4 n-tiles x 512 halves (cols 56..63 zero).
  for (int idx = tid; idx < 8 * 4 * 512; idx += 256) {
    int e = idx & 15, lane = (idx >> 4) & 31, frag = idx >> 9;
    int nt = frag & 3, ks = frag >> 2;
    int k = ks * 32 + (lane >> 4) * 16 + e;     // 0..255
    int n = nt * 16 + (lane & 15);              // 0..63
    float v = (n < 56) ? enc_w[((size_t)g * 56 + n) * 256 + k] : 0.f;
    encBs[idx] = (_Float16)v;
  }

  // Per-layer 64x64 matrices: 2 k-steps x 4 n-tiles x 512 halves each.
  for (int l = 0; l < NL; ++l) {
    for (int idx = tid; idx < 4096; idx += 256) {
      int e = idx & 15, lane = (idx >> 4) & 31, frag = idx >> 9;
      int nt = frag & 3, ks = frag >> 2;
      int k = ks * 32 + (lane >> 4) * 16 + e;   // 0..63
      int n = nt * 16 + (lane & 15);            // 0..63
      // Bcat[k=h][n]: n<32 -> Re(Bbar[n][h]), else Im(Bbar[n-32][h])
      {
        int p = n & 31;
        float br = B_re[((size_t)l * 32 + p) * 64 + k];
        float bi = B_im[((size_t)l * 32 + p) * 64 + k];
        float re = cre[l][p] * br - cim[l][p] * bi;
        float im = cre[l][p] * bi + cim[l][p] * br;
        Bcats[l * 4096 + idx] = (_Float16)(n < 32 ? re : im);
      }
      // WcB[k][n=h]: k<32 -> 2*C_re[h][k], else -2*C_im[h][k-32]
      {
        int pk = k & 31;
        float wv = (k < 32) ? 2.f * C_re[((size_t)l * 64 + n) * 32 + pk]
                            : -2.f * C_im[((size_t)l * 64 + n) * 32 + pk];
        WcBs[l * 4096 + idx] = (_Float16)wv;
      }
      w1Bs[l * 4096 + idx] = (_Float16)glu_w1[(size_t)l * 4096 + n * 64 + k];
      w2Bs[l * 4096 + idx] = (_Float16)glu_w2[(size_t)l * 4096 + n * 64 + k];
    }
  }
}

// ---------------------------------------------------------------- encoder GEMM
__global__ void encoder_kernel(const float* __restrict__ x, const _Float16* __restrict__ encBs,
                               const float* __restrict__ b64, float* __restrict__ h) {
  __shared__ _Float16 tile[ROWS][LDT];
  int wave = threadIdx.x >> 5;
  int lane = threadIdx.x & 31;
  int rows0 = blockIdx.x * ROWS;
  v8f acc[4] = {};
  for (int kc = 0; kc < 256; kc += 64) {
#pragma unroll
    for (int it = 0; it < 8; ++it) {
      int idx = threadIdx.x + 256 * it;       // 0..2047  -> 128x64/4
      int r = idx >> 4;
      int cc = (idx & 15) * 4;
      float4 v = *(const float4*)(x + (size_t)(rows0 + r) * 256 + kc + cc);
      tile[r][cc]     = (_Float16)v.x;
      tile[r][cc + 1] = (_Float16)v.y;
      tile[r][cc + 2] = (_Float16)v.z;
      tile[r][cc + 3] = (_Float16)v.w;
    }
    __syncthreads();
    int ksb = kc >> 5;                        // k-step base (2 per chunk)
    v16h a0 = load_a_frag(&tile[wave * 16][0], LDT, 0);
    v16h a1 = load_a_frag(&tile[wave * 16][0], LDT, 32);
#pragma unroll
    for (int nt = 0; nt < 4; ++nt) {
      acc[nt] = wmma_f16(a0, load_b_frag_swz(encBs, ksb * 4 + nt), acc[nt]);
      acc[nt] = wmma_f16(a1, load_b_frag_swz(encBs, (ksb + 1) * 4 + nt), acc[nt]);
    }
    __syncthreads();
  }
  int nl = lane & 15, mb = (lane >> 4) * 8;
#pragma unroll
  for (int nt = 0; nt < 4; ++nt) {
    int n = nt * 16 + nl;
    float bias = b64[n];
#pragma unroll
    for (int r = 0; r < 8; ++r)
      h[(size_t)(rows0 + wave * 16 + mb + r) * 64 + n] = acc[nt][r] + bias;
  }
}

// ---------------------------------------------------------------- batch stats
__global__ void zero_stats(float* acc) {
  if (threadIdx.x < 128) acc[threadIdx.x] = 0.f;
}

__global__ void stats_kernel(const float* __restrict__ h, float* __restrict__ acc) {
  __shared__ float ssum[4][64], ssq[4][64];
  int c = threadIdx.x & 63;
  int rg = threadIdx.x >> 6;
  size_t base = (size_t)blockIdx.x * CHUNK;
  float s = 0.f, q = 0.f;
  for (int it = 0; it < CHUNK / 4; ++it) {
    float v = h[(base + (size_t)it * 4 + rg) * 64 + c];
    s += v;
    q += v * v;
  }
  ssum[rg][c] = s;
  ssq[rg][c] = q;
  __syncthreads();
  if (threadIdx.x < 64) {
    float ts = ssum[0][c] + ssum[1][c] + ssum[2][c] + ssum[3][c];
    float tq = ssq[0][c] + ssq[1][c] + ssq[2][c] + ssq[3][c];
    atomicAdd(&acc[c], ts);
    atomicAdd(&acc[64 + c], tq);
  }
}

__global__ void finalize_stats(const float* acc, float* stats) {
  int c = threadIdx.x;
  if (c >= 64) return;
  float inv = 1.f / (float)T_LEN;
  float mean = acc[c] * inv;
  float var = acc[64 + c] * inv - mean * mean;
  stats[c] = mean;
  stats[64 + c] = rsqrtf(var + 1e-5f);
}

// -------------------------------------------- norm + Bu = hn @ Bbar_cat^T GEMM
__global__ void bu_gemm_kernel(const float* __restrict__ h, float* __restrict__ hn,
                               float* __restrict__ bu, const _Float16* __restrict__ Bcats,
                               const float* __restrict__ stats,
                               const float* __restrict__ scale, const float* __restrict__ bias) {
  __shared__ _Float16 tile[ROWS][LDT];
  int rows0 = blockIdx.x * ROWS;
#pragma unroll
  for (int it = 0; it < 8; ++it) {
    int idx = threadIdx.x + 256 * it;
    int r = idx >> 4;
    int cc = (idx & 15) * 4;
    float4 v = *(const float4*)(h + (size_t)(rows0 + r) * 64 + cc);
    float o[4] = {v.x, v.y, v.z, v.w};
    float4 hv;
    float* hvp = &hv.x;
#pragma unroll
    for (int j = 0; j < 4; ++j) {
      int c = cc + j;
      float val = (o[j] - stats[c]) * stats[64 + c] * scale[c] + bias[c];
      hvp[j] = val;
      tile[r][c] = (_Float16)val;
    }
    *(float4*)(hn + (size_t)(rows0 + r) * 64 + cc) = hv;
  }
  __syncthreads();
  int wave = threadIdx.x >> 5;
  int lane = threadIdx.x & 31;
  v16h a0 = load_a_frag(&tile[wave * 16][0], LDT, 0);
  v16h a1 = load_a_frag(&tile[wave * 16][0], LDT, 32);
  int nl = lane & 15, mb = (lane >> 4) * 8;
#pragma unroll
  for (int nt = 0; nt < 4; ++nt) {
    v8f acc = {};
    acc = wmma_f16(a0, load_b_frag_swz(Bcats, nt), acc);
    acc = wmma_f16(a1, load_b_frag_swz(Bcats, 4 + nt), acc);
    int n = nt * 16 + nl;
#pragma unroll
    for (int r = 0; r < 8; ++r)
      bu[(size_t)(rows0 + wave * 16 + mb + r) * 64 + n] = acc[r];
  }
}

// ---------------------------------------------------------------- chunked scan
__global__ void scan_local(float* __restrict__ bu, float* __restrict__ carry,
                           const float* __restrict__ LbarT, int layer) {
  int p = threadIdx.x;                      // 0..31 : complex channel
  float lr = LbarT[layer * 64 + p];
  float li = LbarT[layer * 64 + 32 + p];
  float* row = bu + (size_t)blockIdx.x * CHUNK * 64;
  float xr = 0.f, xi = 0.f;
  for (int i = 0; i < CHUNK; ++i, row += 64) {
    __builtin_prefetch(row + 64 * 8, 0, 0);     // global_prefetch_b8
    float br = row[p], bi = row[32 + p];
    float nr = lr * xr - li * xi + br;
    float ni = lr * xi + li * xr + bi;
    xr = nr;
    xi = ni;
    row[p] = xr;
    row[32 + p] = xi;
  }
  carry[blockIdx.x * 64 + p] = xr;
  carry[blockIdx.x * 64 + 32 + p] = xi;
}

__global__ void scan_carry(const float* __restrict__ carry, float* __restrict__ pre,
                           const float* __restrict__ LbarT, int layer) {
  int p = threadIdx.x;
  float lr = LbarT[layer * 64 + p];
  float li = LbarT[layer * 64 + 32 + p];
  // decay = Lbar^CHUNK  (CHUNK = 2^9 -> 9 complex squarings)
  float dr = lr, di = li;
  for (int s = 0; s < 9; ++s) {
    float nr = dr * dr - di * di;
    di = 2.f * dr * di;
    dr = nr;
  }
  float ar = 0.f, ai = 0.f;
  for (int c = 0; c < NCHUNK; ++c) {
    pre[c * 64 + p] = ar;
    pre[c * 64 + 32 + p] = ai;
    float cr = carry[c * 64 + p], ci = carry[c * 64 + 32 + p];
    float nr = dr * ar - di * ai + cr;
    float ni = dr * ai + di * ar + ci;
    ar = nr;
    ai = ni;
  }
}

__global__ void scan_fix(float* __restrict__ bu, const float* __restrict__ pre,
                         const float* __restrict__ LbarT, int layer) {
  int c = blockIdx.x;
  if (c == 0) return;                       // prefix is zero
  int p = threadIdx.x;
  float lr = LbarT[layer * 64 + p];
  float li = LbarT[layer * 64 + 32 + p];
  float ar = pre[c * 64 + p], ai = pre[c * 64 + 32 + p];
  float* row = bu + (size_t)c * CHUNK * 64;
  for (int i = 0; i < CHUNK; ++i, row += 64) {
    __builtin_prefetch(row + 64 * 8, 1, 0);
    float nr = lr * ar - li * ai;           // a <- Lbar^{i+1} * pre
    float ni = lr * ai + li * ar;
    ar = nr;
    ai = ni;
    row[p] += ar;
    row[32 + p] += ai;
  }
}

// ------------------- y = 2Re(xs C^T) + hn*D ; GELU ; GLU ; residual (fused)
__global__ void post_kernel(const float* __restrict__ xs, const float* __restrict__ hn,
                            float* __restrict__ h, const _Float16* __restrict__ WcBs,
                            const _Float16* __restrict__ w1Bs, const _Float16* __restrict__ w2Bs,
                            const float* __restrict__ Dl, const float* __restrict__ b1,
                            const float* __restrict__ b2) {
  __shared__ _Float16 tX[ROWS][LDT];
  __shared__ _Float16 tG[ROWS][LDT];
  int rows0 = blockIdx.x * ROWS;
#pragma unroll
  for (int it = 0; it < 8; ++it) {
    int idx = threadIdx.x + 256 * it;
    int r = idx >> 4;
    int cc = (idx & 15) * 4;
    float4 v = *(const float4*)(xs + (size_t)(rows0 + r) * 64 + cc);
    tX[r][cc]     = (_Float16)v.x;
    tX[r][cc + 1] = (_Float16)v.y;
    tX[r][cc + 2] = (_Float16)v.z;
    tX[r][cc + 3] = (_Float16)v.w;
  }
  __syncthreads();
  int wave = threadIdx.x >> 5;
  int lane = threadIdx.x & 31;
  int m0 = wave * 16;
  int nl = lane & 15, mb = (lane >> 4) * 8;
  v16h a0 = load_a_frag(&tX[m0][0], LDT, 0);
  v16h a1 = load_a_frag(&tX[m0][0], LDT, 32);
#pragma unroll
  for (int nt = 0; nt < 4; ++nt) {
    v8f acc = {};
    acc = wmma_f16(a0, load_b_frag_swz(WcBs, nt), acc);
    acc = wmma_f16(a1, load_b_frag_swz(WcBs, 4 + nt), acc);
    int n = nt * 16 + nl;
    float d = Dl[n];
#pragma unroll
    for (int r = 0; r < 8; ++r) {
      int rl = m0 + mb + r;
      float y = acc[r] + hn[(size_t)(rows0 + rl) * 64 + n] * d;
      tG[rl][n] = (_Float16)gelu_f(y);
    }
  }
  __syncthreads();
  v16h g0 = load_a_frag(&tG[m0][0], LDT, 0);
  v16h g1 = load_a_frag(&tG[m0][0], LDT, 32);
#pragma unroll
  for (int nt = 0; nt < 4; ++nt) {
    v8f u = {}, s = {};
    u = wmma_f16(g0, load_b_frag_swz(w1Bs, nt), u);
    u = wmma_f16(g1, load_b_frag_swz(w1Bs, 4 + nt), u);
    s = wmma_f16(g0, load_b_frag_swz(w2Bs, nt), s);
    s = wmma_f16(g1, load_b_frag_swz(w2Bs, 4 + nt), s);
    int n = nt * 16 + nl;
    float bb1 = b1[n], bb2 = b2[n];
#pragma unroll
    for (int r = 0; r < 8; ++r) {
      size_t off = (size_t)(rows0 + m0 + mb + r) * 64 + n;
      h[off] = h[off] + (u[r] + bb1) * sigmoid_f(s[r] + bb2);
    }
  }
}

// ---------------------------------------------------------------- decoder
__global__ void decoder_kernel(const float* __restrict__ h, const float* __restrict__ dec_w,
                               const float* __restrict__ dec_b, float* __restrict__ out) {
  int t = blockIdx.x * blockDim.x + threadIdx.x;
  if (t >= T_LEN) return;
  const float4* hr = (const float4*)(h + (size_t)t * 64);
  float a0 = dec_b[0], a1 = dec_b[1];
#pragma unroll
  for (int j = 0; j < 16; ++j) {
    float4 v = hr[j];
    const float* w0 = dec_w + j * 4;
    const float* w1 = dec_w + 64 + j * 4;
    a0 += v.x * w0[0] + v.y * w0[1] + v.z * w0[2] + v.w * w0[3];
    a1 += v.x * w1[0] + v.y * w1[1] + v.z * w1[2] + v.w * w1[3];
  }
  out[t * 2] = a0;
  out[t * 2 + 1] = a1;
}

// ---------------------------------------------------------------- launch
extern "C" void kernel_launch(void* const* d_in, const int* in_sizes, int n_in,
                              void* d_out, int out_size, void* d_ws, size_t ws_size,
                              hipStream_t stream) {
  const float* x        = (const float*)d_in[0];
  const int*   gidx     = (const int*)d_in[1];
  const float* enc_w    = (const float*)d_in[2];
  const float* enc_b    = (const float*)d_in[3];
  const float* ctx_emb  = (const float*)d_in[4];
  const float* Lre      = (const float*)d_in[5];
  const float* Lim      = (const float*)d_in[6];
  const float* B_re     = (const float*)d_in[7];
  const float* B_im     = (const float*)d_in[8];
  const float* C_re     = (const float*)d_in[9];
  const float* C_im     = (const float*)d_in[10];
  const float* Dm       = (const float*)d_in[11];
  const float* log_step = (const float*)d_in[12];
  const float* n_scale  = (const float*)d_in[13];
  const float* n_bias   = (const float*)d_in[14];
  const float* glu_w1   = (const float*)d_in[15];
  const float* glu_b1   = (const float*)d_in[16];
  const float* glu_w2   = (const float*)d_in[17];
  const float* glu_b2   = (const float*)d_in[18];
  const float* dec_w    = (const float*)d_in[19];
  const float* dec_b    = (const float*)d_in[20];
  float* out = (float*)d_out;

  char* ws = (char*)d_ws;
  size_t off = 0;
  auto alloc = [&](size_t bytes) -> char* {
    char* p = ws + off;
    off += (bytes + 255) & ~(size_t)255;
    return p;
  };
  float*    h     = (float*)alloc((size_t)T_LEN * 64 * 4);
  float*    hn    = (float*)alloc((size_t)T_LEN * 64 * 4);
  float*    bu    = (float*)alloc((size_t)T_LEN * 64 * 4);
  float*    carry = (float*)alloc((size_t)NCHUNK * 64 * 4);
  float*    pre   = (float*)alloc((size_t)NCHUNK * 64 * 4);
  float*    acc   = (float*)alloc(128 * 4);
  float*    stats = (float*)alloc(128 * 4);
  _Float16* encBs = (_Float16*)alloc(8 * 4 * 512 * 2);
  float*    b64   = (float*)alloc(64 * 4);
  _Float16* Bcats = (_Float16*)alloc(NL * 4096 * 2);
  _Float16* WcBs  = (_Float16*)alloc(NL * 4096 * 2);
  _Float16* w1Bs  = (_Float16*)alloc(NL * 4096 * 2);
  _Float16* w2Bs  = (_Float16*)alloc(NL * 4096 * 2);
  float*    LbarT = (float*)alloc(NL * 64 * 4);

  prep_kernel<<<1, 256, 0, stream>>>(gidx, enc_w, enc_b, ctx_emb, Lre, Lim, B_re, B_im,
                                     C_re, C_im, log_step, glu_w1, glu_w2,
                                     encBs, b64, Bcats, WcBs, w1Bs, w2Bs, LbarT);
  encoder_kernel<<<T_LEN / ROWS, 256, 0, stream>>>(x, encBs, b64, h);

  for (int l = 0; l < NL; ++l) {
    zero_stats<<<1, 128, 0, stream>>>(acc);
    stats_kernel<<<T_LEN / CHUNK, 256, 0, stream>>>(h, acc);
    finalize_stats<<<1, 64, 0, stream>>>(acc, stats);
    bu_gemm_kernel<<<T_LEN / ROWS, 256, 0, stream>>>(h, hn, bu, Bcats + l * 4096, stats,
                                                     n_scale + l * 64, n_bias + l * 64);
    scan_local<<<NCHUNK, 32, 0, stream>>>(bu, carry, LbarT, l);
    scan_carry<<<1, 32, 0, stream>>>(carry, pre, LbarT, l);
    scan_fix<<<NCHUNK, 32, 0, stream>>>(bu, pre, LbarT, l);
    post_kernel<<<T_LEN / ROWS, 256, 0, stream>>>(bu, hn, h, WcBs + l * 4096,
                                                  w1Bs + l * 4096, w2Bs + l * 4096,
                                                  Dm + l * 64, glu_b1 + l * 64,
                                                  glu_b2 + l * 64);
  }
  decoder_kernel<<<T_LEN / 256, 256, 0, stream>>>(h, dec_w, dec_b, out);
}
